// Galerkin_integral_18476949307898
// MI455X (gfx1250) — matrix operand order repaired and checked by reading, not compiled, not run on hardware
//
#include <hip/hip_runtime.h>
#include <hip/hip_bf16.h>

// Problem constants (fixed by the reference: B=8, C=128, H=W=128, heads=8, d=16)
#define NPIX  16384          // H*W
#define CCH   128
#define BIMG  8
#define NHEAD 8
#define DH    16
#define MROWS (BIMG * NPIX)  // 131072 rows of the flattened activation matrix
#define LN_EPS 1e-5f

typedef __attribute__((ext_vector_type(8)))  __bf16 v8bf;
typedef __attribute__((ext_vector_type(16))) __bf16 v16bf;
typedef __attribute__((ext_vector_type(8)))  float  v8f;

static __device__ __forceinline__ v8f wmma_bf16(v16bf a, v16bf b, v8f c) {
  // D = A(16x32 bf16) * B(32x16 bf16) + C(16x16 f32)
  return __builtin_amdgcn_wmma_f32_16x16x32_bf16(false, a, false, b, (short)0, c, false, false);
}

// A-matrix 16x32 bf16 fragment (ISA 7.12.2): lane L holds row m=L%16,
// halves K = (L/16)*8 + 0..7 (VGPR0-3) and K = 16 + (L/16)*8 + 0..7 (VGPR4-7).
// A is row-major with leading dim lda -> two contiguous 16B loads per lane.
static __device__ __forceinline__ v16bf load_a_frag(const __bf16* __restrict__ A,
                                                    size_t lda, int m0, int k0, int lane) {
  int m    = m0 + (lane & 15);
  int koff = k0 + ((lane >> 4) << 3);
  const __bf16* base = A + (size_t)m * lda + koff;
  v8bf lo = *reinterpret_cast<const v8bf*>(base);
  v8bf hi = *reinterpret_cast<const v8bf*>(base + 16);
  v16bf r;
#pragma unroll
  for (int i = 0; i < 8; ++i) { r[i] = lo[i]; r[i + 8] = hi[i]; }
  return r;
}

// B-matrix 32x16 bf16 fragment: lane L holds column n=L%16,
// K = (L/16)*8 + 0..7 and K = 16 + (L/16)*8 + 0..7. Wt is the K-contiguous view of B's columns
// (i.e. weight stored [out,in]) -> two contiguous 16B loads per lane.
static __device__ __forceinline__ v16bf load_b_frag(const __bf16* __restrict__ Wt,
                                                    size_t ldw, int n0, int k0, int lane) {
  int n  = n0 + (lane & 15);
  int ks = k0 + ((lane >> 4) << 3);
  const __bf16* base = Wt + (size_t)n * ldw + ks;
  v8bf lo = *reinterpret_cast<const v8bf*>(base);
  v8bf hi = *reinterpret_cast<const v8bf*>(base + 16);
  v16bf r;
#pragma unroll
  for (int i = 0; i < 8; ++i) { r[i] = lo[i]; r[i + 8] = hi[i]; }
  return r;
}

static __device__ __forceinline__ float gelu_exact(float x) {
  return 0.5f * x * (1.0f + erff(x * 0.70710678118654752f));
}

// ---------------- prep kernels ----------------

__global__ void cvt_bf16_kernel(const float* __restrict__ s, __bf16* __restrict__ d, int n) {
  int i = blockIdx.x * blockDim.x + threadIdx.x;
  if (i < n) d[i] = (__bf16)s[i];
}

// x [B,C,N] f32 -> xt [B,N,C] bf16 (so QKV GEMM A-fragments are K-contiguous)
__global__ void transpose_x_kernel(const float* __restrict__ x, __bf16* __restrict__ xt) {
  size_t i = (size_t)blockIdx.x * blockDim.x + threadIdx.x;  // enumerates x linearly
  int b = (int)(i >> 21);            // C*N = 2^21
  int c = (int)((i >> 14) & 127);    // N = 2^14
  int n = (int)(i & (NPIX - 1));
  xt[(((size_t)b << 14) + (size_t)n) * CCH + c] = (__bf16)x[i];
}

// ---------------- QKV GEMM: [M,128] x [128,384], split into q/k/v per head ----------------
// one wave = one 16x64 output strip (4 col tiles): A fragment reused 4x from registers.
// col tile t: head=t/3, which=t%3 (q/k/v)
__global__ void qkv_gemm_kernel(const __bf16* __restrict__ xt, const __bf16* __restrict__ wqkv,
                                const float* __restrict__ qkv_b,
                                __bf16* __restrict__ qraw, __bf16* __restrict__ kraw,
                                __bf16* __restrict__ vraw) {
  int wave = threadIdx.x >> 5, lane = threadIdx.x & 31;
  int gtile = blockIdx.x * 8 + wave;     // 8192 row tiles * 6 col groups
  int mt = gtile / 6;
  int cg = gtile % 6;                    // group of 4 consecutive 16-col tiles
  int row0 = mt << 4;
  const v8f vzero = {0.f, 0.f, 0.f, 0.f, 0.f, 0.f, 0.f, 0.f};
  v8f acc[4];
#pragma unroll
  for (int j = 0; j < 4; ++j) acc[j] = vzero;
#pragma unroll
  for (int k0 = 0; k0 < 128; k0 += 32) {
    v16bf a = load_a_frag(xt, CCH, row0, k0, lane);
#pragma unroll
    for (int j = 0; j < 4; ++j) {
      v16bf b = load_b_frag(wqkv, CCH, (cg * 4 + j) << 4, k0, lane);
      acc[j] = wmma_bf16(a, b, acc[j]);
    }
  }
  int d = lane & 15, g = lane >> 4;
  int bimg = row0 >> 14;
  int nbase = row0 & (NPIX - 1);
#pragma unroll
  for (int j = 0; j < 4; ++j) {
    int t = cg * 4 + j;
    float bias = qkv_b[(t << 4) + d];
    int head = t / 3, which = t % 3;
    __bf16* dst = (which == 0) ? qraw : ((which == 1) ? kraw : vraw);
    size_t base = ((size_t)(bimg * NHEAD + head) * NPIX + (size_t)nbase) * DH + (size_t)d;
#pragma unroll
    for (int r = 0; r < 8; ++r) {
      int mrow = r + (g << 3);
      dst[base + (size_t)mrow * DH] = (__bf16)(acc[j][r] + bias);
    }
  }
}

// ---------------- LayerNorm over d=16 (ddof=1) + transpose to [bh,16,N] ----------------
__global__ void lnT_kernel(const __bf16* __restrict__ kraw, const __bf16* __restrict__ vraw,
                           const float* __restrict__ klw, const float* __restrict__ klb,
                           const float* __restrict__ vlw, const float* __restrict__ vlb,
                           __bf16* __restrict__ kT, __bf16* __restrict__ vT) {
  int idx = blockIdx.x * blockDim.x + threadIdx.x;   // 64 * NPIX rows
  int bh = idx >> 14;
  int n  = idx & (NPIX - 1);
  int h  = bh & 7;
#pragma unroll
  for (int s = 0; s < 2; ++s) {
    const __bf16* src = s ? vraw : kraw;
    const float*  w   = s ? vlw : klw;
    const float*  bb  = s ? vlb : klb;
    __bf16*       dst = s ? vT : kT;
    const __bf16* row = src + ((size_t)bh * NPIX + (size_t)n) * DH;
    float vals[16];
    float mean = 0.f;
#pragma unroll
    for (int i = 0; i < 16; ++i) { vals[i] = (float)row[i]; mean += vals[i]; }
    mean *= (1.0f / 16.0f);
    float var = 0.f;
#pragma unroll
    for (int i = 0; i < 16; ++i) { float dv = vals[i] - mean; var += dv * dv; }
    var *= (1.0f / 15.0f);                 // unbiased (ddof=1), matches torch-style LN clone
    float inv = 1.0f / (sqrtf(var) + LN_EPS);
#pragma unroll
    for (int i = 0; i < 16; ++i) {
      float y = w[h * 16 + i] * ((vals[i] - mean) * inv) + bb[h * 16 + i];
      dst[((size_t)bh * DH + (size_t)i) * NPIX + (size_t)n] = (__bf16)y;
    }
  }
}

// ---------------- kv = k^T v / N  per (b,h); 16 waves reduce over N via WMMA + LDS ----------------
__global__ void kv_kernel(const __bf16* __restrict__ kT, const __bf16* __restrict__ vT,
                          __bf16* __restrict__ kvT) {
  __shared__ float red[16][256];
  int bh = blockIdx.x;                    // 64
  int wave = threadIdx.x >> 5, lane = threadIdx.x & 31;
  const __bf16* kTb = kT + (size_t)bh * DH * NPIX;
  const __bf16* vTb = vT + (size_t)bh * DH * NPIX;
  v8f acc = {0.f, 0.f, 0.f, 0.f, 0.f, 0.f, 0.f, 0.f};
  for (int i = 0; i < 32; ++i) {
    int n0 = ((wave << 5) + i) << 5;      // 32-wide K chunks over N=16384
    v16bf a = load_a_frag(kTb, NPIX, 0, n0, lane);   // A[m=d, k=n]
    v16bf b = load_b_frag(vTb, NPIX, 0, n0, lane);   // B[k=n, col=e]
    acc = wmma_bf16(a, b, acc);
  }
#pragma unroll
  for (int r = 0; r < 8; ++r) red[wave][lane * 8 + r] = acc[r];
  __syncthreads();
  if (wave == 0) {
    const float invN = 1.0f / (float)NPIX;
    int e = lane & 15, g = lane >> 4;
#pragma unroll
    for (int r = 0; r < 8; ++r) {
      float s = 0.f;
#pragma unroll
      for (int w = 0; w < 16; ++w) s += red[w][lane * 8 + r];
      int dd = r + (g << 3);
      // store transposed [e][d] so av's B-fragment reads are contiguous
      kvT[(size_t)bh * 256 + (size_t)e * 16 + (size_t)dd] = (__bf16)(s * invN);
    }
  }
}

// ---------------- av = q * kv  (K=16 zero-padded to 32), + bias0, -> ret bf16 [B,N,C] ----------------
__global__ void av_kernel(const __bf16* __restrict__ q, const __bf16* __restrict__ kvT,
                          const float* __restrict__ x, __bf16* __restrict__ ret) {
  int wave = threadIdx.x >> 5, lane = threadIdx.x & 31;
  int tile = blockIdx.x * 8 + wave;       // 65536 tiles
  int bh = tile >> 10;
  int n0 = (tile & 1023) << 4;
  int bimg = bh >> 3, h = bh & 7;
  // A fragment: K=0..15 real (q row), K=16..31 zero
  int m    = n0 + (lane & 15);
  int koff = (lane >> 4) << 3;            // 0 or 8, always < 16
  v8bf alo = *reinterpret_cast<const v8bf*>(q + ((size_t)bh * NPIX + (size_t)m) * DH + koff);
  v16bf a;
#pragma unroll
  for (int i = 0; i < 8; ++i) { a[i] = alo[i]; a[i + 8] = (__bf16)0.0f; }
  // B fragment: lanes 0-15 carry K=0..7, lanes 16-31 carry K=8..15 in elements 0-7; elements 8-15 are the zero pad
  const __bf16* kb = kvT + (size_t)bh * 256 + (size_t)(lane & 15) * 16;
  v8bf b0 = reinterpret_cast<const v8bf*>(kb)[0];
  v8bf b1 = reinterpret_cast<const v8bf*>(kb)[1];
  bool hi = (lane >= 16);
  v16bf bfrag;
#pragma unroll
  for (int i = 0; i < 8; ++i) {
    bfrag[i]     = hi ? b1[i] : b0[i];
    bfrag[i + 8] = (__bf16)0.0f;
  }
  v8f acc = {0.f, 0.f, 0.f, 0.f, 0.f, 0.f, 0.f, 0.f};
  acc = wmma_bf16(a, bfrag, acc);
  // ret[b,n,c] = av + x[b,c,n]; residual read is contiguous 8 floats per lane
  int d = lane & 15, g = lane >> 4;
  int cch = h * 16 + d;
  int nb  = n0 + (g << 3);
  const float* xp = x + ((size_t)bimg * CCH + (size_t)cch) * NPIX + (size_t)nb;
#pragma unroll
  for (int r = 0; r < 8; ++r) {
    float v = acc[r] + xp[r];
    ret[((size_t)bimg * NPIX + (size_t)(nb + r)) * CCH + (size_t)cch] = (__bf16)v;
  }
}

// ---------------- o1: GEMM (16x64 per wave) + bias + exact GELU -> h1 bf16 ----------------
__global__ void o1_gemm_kernel(const __bf16* __restrict__ A, const __bf16* __restrict__ W,
                               const float* __restrict__ bias, __bf16* __restrict__ H1) {
  int wave = threadIdx.x >> 5, lane = threadIdx.x & 31;
  int gtile = blockIdx.x * 8 + wave;      // 8192 row tiles * 2 col groups
  int mt = gtile >> 1, cg = gtile & 1;
  int row0 = mt << 4;
  const v8f vzero = {0.f, 0.f, 0.f, 0.f, 0.f, 0.f, 0.f, 0.f};
  v8f acc[4];
#pragma unroll
  for (int j = 0; j < 4; ++j) acc[j] = vzero;
#pragma unroll
  for (int k0 = 0; k0 < 128; k0 += 32) {
    v16bf a = load_a_frag(A, CCH, row0, k0, lane);
#pragma unroll
    for (int j = 0; j < 4; ++j) {
      v16bf b = load_b_frag(W, CCH, (cg * 4 + j) << 4, k0, lane);
      acc[j] = wmma_bf16(a, b, acc[j]);
    }
  }
  int d = lane & 15, g = lane >> 4;
#pragma unroll
  for (int j = 0; j < 4; ++j) {
    int ccol = ((cg * 4 + j) << 4) + d;
    float bv = bias[ccol];
#pragma unroll
    for (int r = 0; r < 8; ++r) {
      int mrow = r + (g << 3);
      float v = gelu_exact(acc[j][r] + bv);
      H1[(size_t)(row0 + mrow) * CCH + (size_t)ccol] = (__bf16)v;
    }
  }
}

// ---------------- o2: GEMM (16x64 per wave) + bias + bias0(x) -> out f32 [B,C,N] ----------------
__global__ void o2_gemm_kernel(const __bf16* __restrict__ A, const __bf16* __restrict__ W,
                               const float* __restrict__ bias, const float* __restrict__ x,
                               float* __restrict__ out) {
  int wave = threadIdx.x >> 5, lane = threadIdx.x & 31;
  int gtile = blockIdx.x * 8 + wave;      // 8192 row tiles * 2 col groups
  int mt = gtile >> 1, cg = gtile & 1;
  int row0 = mt << 4;
  const v8f vzero = {0.f, 0.f, 0.f, 0.f, 0.f, 0.f, 0.f, 0.f};
  v8f acc[4];
#pragma unroll
  for (int j = 0; j < 4; ++j) acc[j] = vzero;
#pragma unroll
  for (int k0 = 0; k0 < 128; k0 += 32) {
    v16bf a = load_a_frag(A, CCH, row0, k0, lane);
#pragma unroll
    for (int j = 0; j < 4; ++j) {
      v16bf b = load_b_frag(W, CCH, (cg * 4 + j) << 4, k0, lane);
      acc[j] = wmma_bf16(a, b, acc[j]);
    }
  }
  int d = lane & 15, g = lane >> 4;
  int b = row0 >> 14;
  int nbase = (row0 & (NPIX - 1)) + (g << 3);
#pragma unroll
  for (int j = 0; j < 4; ++j) {
    int cch = ((cg * 4 + j) << 4) + d;
    float bv = bias[cch];
    size_t off = ((size_t)b * CCH + (size_t)cch) * NPIX + (size_t)nbase;
    const float4* xp = reinterpret_cast<const float4*>(x + off);
    float4 xa = xp[0], xb = xp[1];
    float4 oa, ob;
    oa.x = acc[j][0] + bv + xa.x; oa.y = acc[j][1] + bv + xa.y;
    oa.z = acc[j][2] + bv + xa.z; oa.w = acc[j][3] + bv + xa.w;
    ob.x = acc[j][4] + bv + xb.x; ob.y = acc[j][5] + bv + xb.y;
    ob.z = acc[j][6] + bv + xb.z; ob.w = acc[j][7] + bv + xb.w;
    float4* op = reinterpret_cast<float4*>(out + off);
    op[0] = oa; op[1] = ob;
  }
}

extern "C" void kernel_launch(void* const* d_in, const int* in_sizes, int n_in,
                              void* d_out, int out_size, void* d_ws, size_t ws_size,
                              hipStream_t stream) {
  (void)in_sizes; (void)n_in; (void)out_size; (void)ws_size;
  const float* x     = (const float*)d_in[0];
  const float* qkv_w = (const float*)d_in[1];
  const float* qkv_b = (const float*)d_in[2];
  const float* o1_w  = (const float*)d_in[3];
  const float* o1_b  = (const float*)d_in[4];
  const float* o2_w  = (const float*)d_in[5];
  const float* o2_b  = (const float*)d_in[6];
  const float* klw   = (const float*)d_in[7];
  const float* klb   = (const float*)d_in[8];
  const float* vlw   = (const float*)d_in[9];
  const float* vlb   = (const float*)d_in[10];
  float* out = (float*)d_out;

  char* ws = (char*)d_ws;
  size_t off = 0;
  auto alloc = [&](size_t bytes) -> void* {
    void* p = ws + off;
    off += (bytes + 255) & ~(size_t)255;
    return p;
  };
  __bf16* xt   = (__bf16*)alloc((size_t)MROWS * CCH * 2);        // 33.5 MB
  __bf16* wqkv = (__bf16*)alloc((size_t)384 * 128 * 2);
  __bf16* w1   = (__bf16*)alloc((size_t)128 * 128 * 2);
  __bf16* w2   = (__bf16*)alloc((size_t)128 * 128 * 2);
  __bf16* qraw = (__bf16*)alloc((size_t)64 * NPIX * DH * 2);     // 33.5 MB
  __bf16* kraw = (__bf16*)alloc((size_t)64 * NPIX * DH * 2);     // 33.5 MB
  __bf16* vraw = (__bf16*)alloc((size_t)64 * NPIX * DH * 2);     // 33.5 MB
  __bf16* kT   = (__bf16*)alloc((size_t)64 * DH * NPIX * 2);     // 33.5 MB
  __bf16* vT   = (__bf16*)alloc((size_t)64 * DH * NPIX * 2);     // 33.5 MB
  __bf16* kvT  = (__bf16*)alloc((size_t)64 * 16 * 16 * 2);
  // kraw/vraw are dead after lnT_kernel: reuse for ret / h1 (same size)
  __bf16* retb = kraw;
  __bf16* h1b  = vraw;

  cvt_bf16_kernel<<<(384 * 128 + 255) / 256, 256, 0, stream>>>(qkv_w, wqkv, 384 * 128);
  cvt_bf16_kernel<<<(128 * 128 + 255) / 256, 256, 0, stream>>>(o1_w, w1, 128 * 128);
  cvt_bf16_kernel<<<(128 * 128 + 255) / 256, 256, 0, stream>>>(o2_w, w2, 128 * 128);
  transpose_x_kernel<<<(unsigned)(((size_t)MROWS * CCH) / 256), 256, 0, stream>>>(x, xt);
  qkv_gemm_kernel<<<6144, 256, 0, stream>>>(xt, wqkv, qkv_b, qraw, kraw, vraw);
  lnT_kernel<<<(64 * NPIX) / 256, 256, 0, stream>>>(kraw, vraw, klw, klb, vlw, vlb, kT, vT);
  kv_kernel<<<64, 512, 0, stream>>>(kT, vT, kvT);
  av_kernel<<<8192, 256, 0, stream>>>(qraw, kvT, x, retb);
  o1_gemm_kernel<<<2048, 256, 0, stream>>>(retb, w1, o1_b, h1b);
  o2_gemm_kernel<<<2048, 256, 0, stream>>>(h1b, w2, o2_b, x, out);
}
